// JointNEDynamicKGNN_21251498180618
// MI455X (gfx1250) — matrix operand order repaired and checked by reading, not compiled, not run on hardware
//
#include <hip/hip_runtime.h>
#include <hip/hip_bf16.h>

typedef __attribute__((ext_vector_type(16))) __bf16 v16bf;
typedef __attribute__((ext_vector_type(8)))  __bf16 v8bf;
typedef __attribute__((ext_vector_type(8)))  float  v8f;

#define HDIM    256
#define NNODES  100000
#define NEDGES  800000
#define NLAYERS 3
#define NGRAPH  64

// ---------------- utility kernels ----------------

__global__ void k_zero(float* __restrict__ p, int n) {
  int i = blockIdx.x * blockDim.x + threadIdx.x;
  if (i < n) p[i] = 0.f;
}

__global__ void k_deg(const int* __restrict__ dst, float* __restrict__ deg, int e) {
  int i = blockIdx.x * blockDim.x + threadIdx.x;
  if (i < e) atomicAdd(&deg[dst[i]], 1.f);
}

// in[R][C] (f32, row-major) -> out[C][R] (bf16, row-major)
__global__ void k_transcvt(const float* __restrict__ in, __bf16* __restrict__ out,
                           int R, int C) {
  int i = blockIdx.x * blockDim.x + threadIdx.x;
  if (i >= R * C) return;
  int r = i % R, c = i / R;
  out[i] = (__bf16)in[(size_t)r * C + c];   // out[c*R + r]
}

__global__ void k_cvt(const float* __restrict__ in, __bf16* __restrict__ out, int n) {
  int i = blockIdx.x * blockDim.x + threadIdx.x;
  if (i < n) out[i] = (__bf16)in[i];
}

// msg_bf16 = bf16(msg / max(deg,1))  (deg is per node; H=256 features/node)
__global__ void k_cvt_msg(const float* __restrict__ msg, const float* __restrict__ deg,
                          __bf16* __restrict__ out, int n) {
  int i = blockIdx.x * blockDim.x + threadIdx.x;
  if (i >= n) return;
  float d = deg[i >> 8];
  d = d < 1.f ? 1.f : d;
  out[i] = (__bf16)(msg[i] / d);
}

// msg[dst] += h[src], 4 features per thread (64 threads per edge -> coalesced row reads)
__global__ void k_scatter(const float* __restrict__ h, const int* __restrict__ src,
                          const int* __restrict__ dst, float* __restrict__ msg, int e) {
  int i = blockIdx.x * blockDim.x + threadIdx.x;
  if (i >= e * (HDIM / 4)) return;
  int ed = i >> 6;
  int c  = (i & 63) << 2;
  const float4 v = *(const float4*)(h + (size_t)src[ed] * HDIM + c);
  float* out = msg + (size_t)dst[ed] * HDIM + c;
  atomicAdd(out + 0, v.x); atomicAdd(out + 1, v.y);
  atomicAdd(out + 2, v.z); atomicAdd(out + 3, v.w);
}

// ---------------- WMMA GEMM: h_new = relu(hb@Ws + msgb@Wn + bias) ----------------
// Block = 256 threads = 8 waves; block computes 16 rows x 256 cols.
// Wave w covers columns [32w, 32w+32) as two 16x16 tiles. K = 256, step 32.
__global__ __launch_bounds__(256) void k_gemm_layer(
    const __bf16* __restrict__ A,  const __bf16* __restrict__ A2,
    const __bf16* __restrict__ BT, const __bf16* __restrict__ BT2,
    const float* __restrict__ bias, float* __restrict__ out) {
  const int row0 = blockIdx.x * 16;
  const int wave = threadIdx.x >> 5;
  const int lane = threadIdx.x & 31;
  const int m  = lane & 15;
  const int kh = lane >> 4;
  const int n0 = wave * 32;

  v8f acc0 = {}; v8f acc1 = {};
  const __bf16* ar   = A   + (size_t)(row0 + m) * HDIM;
  const __bf16* ar2  = A2  + (size_t)(row0 + m) * HDIM;
  const __bf16* br0  = BT  + (size_t)(n0 + m) * HDIM + kh * 16;
  const __bf16* br1  = BT  + (size_t)(n0 + 16 + m) * HDIM + kh * 16;
  const __bf16* br0b = BT2 + (size_t)(n0 + m) * HDIM + kh * 16;
  const __bf16* br1b = BT2 + (size_t)(n0 + 16 + m) * HDIM + kh * 16;

#pragma unroll
  for (int kb = 0; kb < HDIM; kb += 32) {
    v16bf a, a2;
    ((v8bf*)&a)[0]  = *(const v8bf*)(ar  + kb + kh * 16);      // K = kb+kh*16 .. +7
    ((v8bf*)&a)[1]  = *(const v8bf*)(ar  + kb + kh * 16 + 8);  // K = kb+kh*16+8 .. +15
    ((v8bf*)&a2)[0] = *(const v8bf*)(ar2 + kb + kh * 16);
    ((v8bf*)&a2)[1] = *(const v8bf*)(ar2 + kb + kh * 16 + 8);
    v16bf b0  = *(const v16bf*)(br0  + kb);
    v16bf b1  = *(const v16bf*)(br1  + kb);
    v16bf b0b = *(const v16bf*)(br0b + kb);
    v16bf b1b = *(const v16bf*)(br1b + kb);
    acc0 = __builtin_amdgcn_wmma_f32_16x16x32_bf16(false, a,  false, b0,  (short)0, acc0, false, false);
    acc0 = __builtin_amdgcn_wmma_f32_16x16x32_bf16(false, a2, false, b0b, (short)0, acc0, false, false);
    acc1 = __builtin_amdgcn_wmma_f32_16x16x32_bf16(false, a,  false, b1,  (short)0, acc1, false, false);
    acc1 = __builtin_amdgcn_wmma_f32_16x16x32_bf16(false, a2, false, b1b, (short)0, acc1, false, false);
  }

  const int col0 = n0 + m, col1 = n0 + 16 + m;
  const float bf0 = bias[col0], bf1 = bias[col1];
#pragma unroll
  for (int v = 0; v < 8; ++v) {
    const int row = row0 + kh * 8 + v;
    float x0 = acc0[v] + bf0; x0 = x0 > 0.f ? x0 : 0.f;
    float x1 = acc1[v] + bf1; x1 = x1 > 0.f ? x1 : 0.f;
    out[(size_t)row * HDIM + col0] = x0;
    out[(size_t)row * HDIM + col1] = x1;
  }
}

// ---------------- node head: out = relu(hb@W1 + b1) @ W2 + b2 ----------------
// Block = 256 threads = 8 waves, 16 rows; wave w covers columns [16w,16w+16) of 128.
__global__ __launch_bounds__(256) void k_nodehead(
    const __bf16* __restrict__ A, const __bf16* __restrict__ W1T,
    const float* __restrict__ b1, const float* __restrict__ w2,
    const float* __restrict__ b2, float* __restrict__ out) {
  __shared__ float rowsum[16];
  const int row0 = blockIdx.x * 16;
  const int wave = threadIdx.x >> 5;
  const int lane = threadIdx.x & 31;
  const int m  = lane & 15;
  const int kh = lane >> 4;
  const int n0 = wave * 16;

  v8f acc = {};
  const __bf16* ar = A   + (size_t)(row0 + m) * HDIM;
  const __bf16* br = W1T + (size_t)(n0 + m) * HDIM + kh * 16;
#pragma unroll
  for (int kb = 0; kb < HDIM; kb += 32) {
    v16bf a;
    ((v8bf*)&a)[0] = *(const v8bf*)(ar + kb + kh * 16);
    ((v8bf*)&a)[1] = *(const v8bf*)(ar + kb + kh * 16 + 8);
    v16bf b = *(const v16bf*)(br + kb);
    acc = __builtin_amdgcn_wmma_f32_16x16x32_bf16(false, a, false, b, (short)0, acc, false, false);
  }
  if (threadIdx.x < 16) rowsum[threadIdx.x] = 0.f;
  __syncthreads();
  const int col = n0 + m;
  const float bb = b1[col], ww = w2[col];
#pragma unroll
  for (int v = 0; v < 8; ++v) {
    float t = acc[v] + bb; t = t > 0.f ? t : 0.f; t *= ww;
    t += __shfl_xor(t, 1); t += __shfl_xor(t, 2);
    t += __shfl_xor(t, 4); t += __shfl_xor(t, 8);
    if (m == 0) atomicAdd(&rowsum[kh * 8 + v], t);
  }
  __syncthreads();
  if (threadIdx.x < 16) out[row0 + threadIdx.x] = rowsum[threadIdx.x] + b2[0];
}

// ---------------- attention pooling ----------------

__global__ __launch_bounds__(256) void k_score(const float* __restrict__ h,
                                               const float* __restrict__ aw,
                                               const float* __restrict__ ab,
                                               float* __restrict__ s, int n) {
  int wave = threadIdx.x >> 5, lane = threadIdx.x & 31;
  int node = blockIdx.x * 8 + wave;
  if (node >= n) return;
  float acc = 0.f;
  for (int j = lane; j < HDIM; j += 32) acc += h[(size_t)node * HDIM + j] * aw[j];
  acc += __shfl_xor(acc, 16); acc += __shfl_xor(acc, 8); acc += __shfl_xor(acc, 4);
  acc += __shfl_xor(acc, 2);  acc += __shfl_xor(acc, 1);
  if (lane == 0) s[node] = acc + ab[0];
}

__device__ inline unsigned enc_f(float f) {
  unsigned u = __float_as_uint(f);
  return u ^ ((u & 0x80000000u) ? 0xFFFFFFFFu : 0x80000000u);
}
__device__ inline float dec_f(unsigned e) {
  return __uint_as_float(e ^ ((e & 0x80000000u) ? 0x80000000u : 0xFFFFFFFFu));
}

__global__ void k_gmax(const float* __restrict__ s, const int* __restrict__ batch,
                       unsigned* __restrict__ gmax, int n) {
  int i = blockIdx.x * blockDim.x + threadIdx.x;
  if (i < n) atomicMax(&gmax[batch[i]], enc_f(s[i]));
}

__global__ void k_exp(float* __restrict__ s, const int* __restrict__ batch,
                      const unsigned* __restrict__ gmax, float* __restrict__ denom, int n) {
  int i = blockIdx.x * blockDim.x + threadIdx.x;
  if (i >= n) return;
  int b = batch[i];
  float e = expf(s[i] - dec_f(gmax[b]));
  s[i] = e;
  atomicAdd(&denom[b], e);
}

__global__ void k_pool(const float* __restrict__ h, const float* __restrict__ e,
                       const float* __restrict__ denom, const int* __restrict__ batch,
                       float* __restrict__ g, int n) {
  int i = blockIdx.x * blockDim.x + threadIdx.x;
  if (i >= n * (HDIM / 4)) return;
  int node = i >> 6;
  int c = (i & 63) << 2;
  int b = batch[node];
  float alpha = e[node] / denom[b];
  const float4 v = *(const float4*)(h + (size_t)node * HDIM + c);
  float* out = g + (size_t)b * HDIM + c;
  atomicAdd(out + 0, alpha * v.x); atomicAdd(out + 1, alpha * v.y);
  atomicAdd(out + 2, alpha * v.z); atomicAdd(out + 3, alpha * v.w);
}

// graph head: 64 blocks x 128 threads; thread j = hidden unit
__global__ __launch_bounds__(128) void k_graphhead(
    const float* __restrict__ g, const float* __restrict__ W1,
    const float* __restrict__ b1, const float* __restrict__ W2,
    const float* __restrict__ b2, float* __restrict__ out) {
  __shared__ float ws[4];
  int j = threadIdx.x, gi = blockIdx.x;
  float acc = 0.f;
  for (int k = 0; k < HDIM; ++k) acc += g[(size_t)gi * HDIM + k] * W1[(size_t)k * (HDIM / 2) + j];
  float t = acc + b1[j]; t = t > 0.f ? t : 0.f; t *= W2[j];
  t += __shfl_xor(t, 16); t += __shfl_xor(t, 8); t += __shfl_xor(t, 4);
  t += __shfl_xor(t, 2);  t += __shfl_xor(t, 1);
  if ((j & 31) == 0) ws[j >> 5] = t;
  __syncthreads();
  if (j == 0) out[gi] = ws[0] + ws[1] + ws[2] + ws[3] + b2[0];
}

// ---------------- launch ----------------

extern "C" void kernel_launch(void* const* d_in, const int* in_sizes, int n_in,
                              void* d_out, int out_size, void* d_ws, size_t ws_size,
                              hipStream_t stream) {
  (void)in_sizes; (void)n_in; (void)out_size; (void)ws_size;
  const float* x     = (const float*)d_in[0];
  const int*   ei    = (const int*)d_in[1];
  const int*   src   = ei;
  const int*   dst   = ei + NEDGES;
  const int*   batch = (const int*)d_in[2];
  const float* Ws    = (const float*)d_in[3];
  const float* Wn    = (const float*)d_in[4];
  const float* bs    = (const float*)d_in[5];
  const float* attw  = (const float*)d_in[6];
  const float* attb  = (const float*)d_in[7];
  const float* neW1  = (const float*)d_in[8];
  const float* neb1  = (const float*)d_in[9];
  const float* neW2  = (const float*)d_in[10];
  const float* neb2  = (const float*)d_in[11];
  const float* ndW1  = (const float*)d_in[12];
  const float* ndb1  = (const float*)d_in[13];
  const float* ndW2  = (const float*)d_in[14];
  const float* ndb2  = (const float*)d_in[15];

  char* p = (char*)d_ws;
  auto carve = [&](size_t bytes) -> char* {
    char* r = p; p += (bytes + 255) & ~(size_t)255; return r;
  };
  float*    h     = (float*)carve(sizeof(float) * (size_t)NNODES * HDIM);
  float*    msg   = (float*)carve(sizeof(float) * (size_t)NNODES * HDIM);
  __bf16*   hb    = (__bf16*)carve(2 * (size_t)NNODES * HDIM);
  __bf16*   msgb  = (__bf16*)carve(2 * (size_t)NNODES * HDIM);
  __bf16*   WsT   = (__bf16*)carve(2 * (size_t)NLAYERS * HDIM * HDIM);
  __bf16*   WnT   = (__bf16*)carve(2 * (size_t)NLAYERS * HDIM * HDIM);
  __bf16*   ndW1T = (__bf16*)carve(2 * (size_t)HDIM * (HDIM / 2));
  float*    deg   = (float*)carve(sizeof(float) * NNODES);
  float*    sc    = (float*)carve(sizeof(float) * NNODES);
  unsigned* gmax  = (unsigned*)carve(sizeof(unsigned) * NGRAPH);
  float*    gden  = (float*)carve(sizeof(float) * NGRAPH);
  float*    gp    = (float*)carve(sizeof(float) * NGRAPH * HDIM);

  const int B = 256;
  auto nb = [](long long n, int b) { return (int)((n + b - 1) / b); };
  const int NH = NNODES * HDIM;

  // degrees
  k_zero<<<nb(NNODES, B), B, 0, stream>>>(deg, NNODES);
  k_deg<<<nb(NEDGES, B), B, 0, stream>>>(dst, deg, NEDGES);

  // stage weights: bf16, transposed so B-fragment loads are 32B contiguous
  for (int l = 0; l < NLAYERS; ++l) {
    k_transcvt<<<nb(HDIM * HDIM, B), B, 0, stream>>>(Ws + (size_t)l * HDIM * HDIM,
                                                     WsT + (size_t)l * HDIM * HDIM, HDIM, HDIM);
    k_transcvt<<<nb(HDIM * HDIM, B), B, 0, stream>>>(Wn + (size_t)l * HDIM * HDIM,
                                                     WnT + (size_t)l * HDIM * HDIM, HDIM, HDIM);
  }
  k_transcvt<<<nb(HDIM * (HDIM / 2), B), B, 0, stream>>>(ndW1, ndW1T, HDIM, HDIM / 2);

  // SAGE layers
  const float* hprev = x;
  for (int l = 0; l < NLAYERS; ++l) {
    k_zero<<<nb(NH, B), B, 0, stream>>>(msg, NH);
    k_scatter<<<nb((long long)NEDGES * (HDIM / 4), B), B, 0, stream>>>(hprev, src, dst, msg, NEDGES);
    k_cvt<<<nb(NH, B), B, 0, stream>>>(hprev, hb, NH);
    k_cvt_msg<<<nb(NH, B), B, 0, stream>>>(msg, deg, msgb, NH);
    k_gemm_layer<<<NNODES / 16, 256, 0, stream>>>(hb, msgb,
        WsT + (size_t)l * HDIM * HDIM, WnT + (size_t)l * HDIM * HDIM,
        bs + (size_t)l * HDIM, h);
    hprev = h;
  }

  // final bf16 copy of h for the node head
  k_cvt<<<nb(NH, B), B, 0, stream>>>(h, hb, NH);

  // attention pooling
  k_score<<<nb(NNODES, 8), 256, 0, stream>>>(h, attw, attb, sc, NNODES);
  k_zero<<<1, B, 0, stream>>>((float*)gmax, NGRAPH);   // 0 < enc(-inf): safe init
  k_zero<<<1, B, 0, stream>>>(gden, NGRAPH);
  k_zero<<<nb(NGRAPH * HDIM, B), B, 0, stream>>>(gp, NGRAPH * HDIM);
  k_gmax<<<nb(NNODES, B), B, 0, stream>>>(sc, batch, gmax, NNODES);
  k_exp<<<nb(NNODES, B), B, 0, stream>>>(sc, batch, gmax, gden, NNODES);
  k_pool<<<nb((long long)NNODES * (HDIM / 4), B), B, 0, stream>>>(h, sc, gden, batch, gp, NNODES);

  // heads
  k_graphhead<<<NGRAPH, HDIM / 2, 0, stream>>>(gp, neW1, neb1, neW2, neb2, (float*)d_out);
  k_nodehead<<<NNODES / 16, 256, 0, stream>>>(hb, ndW1T, ndb1, ndW2, ndb2, (float*)d_out + NGRAPH);
}
